// MultiLayerGAT_45277545234768
// MI455X (gfx1250) — matrix operand order repaired and checked by reading, not compiled, not run on hardware
//
#include <hip/hip_runtime.h>

#define GAT_ALPHA 0.2f
#define GAT_NEG_INF -9e15f

typedef __attribute__((ext_vector_type(16))) __bf16 v16bf;
typedef __attribute__((ext_vector_type(8)))  float v8f;

union Frag {
    v16bf v;
    unsigned int u[8];
};

__device__ __forceinline__ unsigned short f2bf(float f) {
    unsigned int u = __float_as_uint(f);
    u += 0x7FFFu + ((u >> 16) & 1u);   // round-to-nearest-even
    return (unsigned short)(u >> 16);
}
__device__ __forceinline__ unsigned int pack2bf(float lo, float hi) {
    return (unsigned int)f2bf(lo) | ((unsigned int)f2bf(hi) << 16);
}
// A-matrix 16x32 bf16 lane layout (ISA 7.12.2): lanes 0-15 M=lane, K chunks 0-7/16-23;
// lanes 16-31 same M, K chunks 8-15/24-31. VGPR v holds K = k0A, k0A+1.
__device__ __forceinline__ int k0A(int v, int hf) {
    return (v < 4 ? 2 * v : 16 + 2 * (v - 4)) + hf * 8;
}
// B-matrix 32x16 bf16: N = lane&15, same K order as A (lanes 0-15 K 0-7/16-23, lanes 16-31 K 8-15/24-31),
// VGPR v holds K = k0A(v, hf), +1.
__device__ __forceinline__ int k0B(int v, int hf) {
    return (v < 4 ? 2 * v : 16 + 2 * (v - 4)) + hf * 8;
}
__device__ __forceinline__ float lrelu(float x) { return x >= 0.f ? x : GAT_ALPHA * x; }
__device__ __forceinline__ float eluf(float x)  { return x > 0.f ? x : __expf(x) - 1.f; }

// Async memory->LDS copy (CDNA5 GLOBAL_LOAD_ASYNC_TO_LDS_B128, tracked by ASYNCcnt).
// lds_off is the 32-bit LDS byte address (low 32 bits of the generic shared pointer).
__device__ __forceinline__ void async_g2l_b128(unsigned lds_off, unsigned long long gaddr) {
    asm volatile("global_load_async_to_lds_b128 %0, %1, off"
                 :: "v"(lds_off), "v"(gaddr) : "memory");
}
__device__ __forceinline__ void wait_asynccnt0() {
    asm volatile("s_wait_asynccnt 0x0" ::: "memory");
}
__device__ __forceinline__ unsigned lds_addr32(const void* p) {
    return (unsigned)(unsigned long long)(size_t)p;
}

// ---------------- elementwise convert / transpose ----------------
__global__ void cvt_bf16_kernel(const float* __restrict__ src,
                                unsigned short* __restrict__ dst, int n) {
    int t = blockIdx.x * blockDim.x + threadIdx.x;
    if (t < n) dst[t] = f2bf(src[t]);
}

// dst[c*R + r] = bf16(src[r*C + c])   (store B transposed so K pairs are contiguous)
__global__ void transpose_bf16_kernel(const float* __restrict__ src,
                                      unsigned short* __restrict__ dst, int R, int C) {
    int t = blockIdx.x * blockDim.x + threadIdx.x;
    if (t < R * C) {
        int c = t / R, r = t % R;
        dst[c * R + r] = f2bf(src[r * C + c]);
    }
}

// ---------------- generic bf16 WMMA GEMM: C[M,N] = A[M,K] * Bt[N,K]^T ----------------
// Also writes Ct[c*M + i] (bf16 transposed copy of C) for the downstream attention GEMM.
__global__ void __launch_bounds__(256)
gemm_bf16_kernel(const unsigned short* __restrict__ A,
                 const unsigned short* __restrict__ Bt,
                 float* __restrict__ C, unsigned short* __restrict__ Ct,
                 int M, int N, int K) {
    int lane = threadIdx.x & 31, wave = threadIdx.x >> 5;
    int hf = lane >> 4, ln = lane & 15;
    int tile = blockIdx.x * 8 + wave;
    int ntj = N >> 4;
    int ti = tile / ntj, tj = tile % ntj;
    int i0 = ti * 16, c0 = tj * 16;

    v8f acc = {0.f, 0.f, 0.f, 0.f, 0.f, 0.f, 0.f, 0.f};
    const unsigned short* arow = A + (size_t)(i0 + ln) * K;
    const unsigned short* brow = Bt + (size_t)(c0 + ln) * K;

    for (int kb = 0; kb < K; kb += 32) {
        Frag a, b;
#pragma unroll
        for (int v = 0; v < 8; ++v) {
            a.u[v] = *(const unsigned int*)(arow + kb + k0A(v, hf));
            b.u[v] = *(const unsigned int*)(brow + kb + k0B(v, hf));
        }
        acc = __builtin_amdgcn_wmma_f32_16x16x32_bf16(false, a.v, false, b.v,
                                                      (short)0, acc, false, false);
    }
#pragma unroll
    for (int r = 0; r < 8; ++r) {
        int i = i0 + hf * 8 + r, c = c0 + ln;
        C[(size_t)i * N + c] = acc[r];
        Ct[(size_t)c * M + i] = f2bf(acc[r]);
    }
}

// ---------------- f_src / f_dst projections ----------------
__global__ void head_proj_kernel(const float* __restrict__ h,
                                 const float* __restrict__ a_src,
                                 const float* __restrict__ a_dst,
                                 float* __restrict__ fsrc, float* __restrict__ fdst,
                                 int N, int H, int F) {
    int t = blockIdx.x * blockDim.x + threadIdx.x;
    if (t >= N * H) return;
    int hh = t / N, n = t % N;
    const float* row = h + (size_t)n * H * F + hh * F;
    const float* as = a_src + hh * F;
    const float* ad = a_dst + hh * F;
    float s = 0.f, d = 0.f;
    for (int o = 0; o < F; ++o) {
        float v = row[o];
        s += v * as[o];
        d += v * ad[o];
    }
    fsrc[hh * N + n] = s;
    fdst[hh * N + n] = d;
}

// ---------------- layer 1 fused attention (8 heads, Fout=128) ----------------
// One block = 16 destination rows; wave w owns head w and produces a 16x128 output panel.
__global__ void __launch_bounds__(256)
attn1_kernel(const float* __restrict__ adj,
             const float* __restrict__ fsrc, const float* __restrict__ fdst,
             const unsigned short* __restrict__ Ht,   // h1^T bf16 [1024][4096]
             float* __restrict__ out_f32,             // elu(out) f32 [4096][1024]
             unsigned short* __restrict__ out_bf16)   // elu(out) bf16 [4096][1024]
{
    __shared__ float fdLds[8][4096];        // per-head fdst staged via async copy (128 KB)
    __shared__ unsigned int maskLds[16 * 128];
    __shared__ float sMax[8][16];
    __shared__ float sInv[8][16];
    const int N = 4096;
    int lane = threadIdx.x & 31, wave = threadIdx.x >> 5;
    int hf = lane >> 4, ln = lane & 15;
    int i0 = blockIdx.x * 16;
    int h = wave;

    // Kick off async staging of this head's fdst into LDS (overlaps with mask build)
    {
        const float* src = fdst + h * N;
        for (int j4 = lane * 4; j4 < 4096; j4 += 128) {
            async_g2l_b128(lds_addr32(&fdLds[h][j4]),
                           (unsigned long long)(size_t)(src + j4));
        }
    }

    // Phase A1: adjacency bitmask (read adj once per block, shared by all heads)
    for (int wi = wave; wi < 2048; wi += 8) {
        int r = wi >> 7, cw = wi & 127;
        float v = adj[(size_t)(i0 + r) * N + cw * 32 + lane];
        unsigned long long b = __ballot(v > 0.f);
        if (lane == 0) maskLds[wi] = (unsigned int)b;
    }
    wait_asynccnt0();
    __syncthreads();

    // Phase A2: online softmax stats; wave w handles head w, all 16 rows
    for (int r = 0; r < 16; ++r) {
        float fs = fsrc[h * N + i0 + r];
        float m = GAT_NEG_INF, s = 0.f;
        for (int cw = 0; cw < 128; ++cw) {
            unsigned int mw = maskLds[r * 128 + cw];
            if (mw == 0u) continue;           // uniform across the wave
            if ((mw >> lane) & 1u) {
                float e = lrelu(fs + fdLds[h][cw * 32 + lane]);
                float M = fmaxf(m, e);
                s = s * __expf(m - M) + __expf(e - M);
                m = M;
            }
        }
#pragma unroll
        for (int off = 16; off >= 1; off >>= 1) {
            float mo = __shfl_xor(m, off, 32);
            float so = __shfl_xor(s, off, 32);
            float M = fmaxf(m, mo);
            s = s * __expf(m - M) + so * __expf(mo - M);
            m = M;
        }
        if (lane == 0) {
            sMax[h][r] = m;
            sInv[h][r] = s > 0.f ? 1.f / s : 0.f;
        }
    }
    __syncthreads();

    // Phase B: out = attn @ h  via WMMA; attention weights built per-lane in A layout
    float fsA = fsrc[h * N + i0 + ln];
    float rmax = sMax[h][ln], rinv = sInv[h][ln];
    int hbase = h * 128;
    v8f zero = {0.f, 0.f, 0.f, 0.f, 0.f, 0.f, 0.f, 0.f};
    v8f acc[8];
#pragma unroll
    for (int t = 0; t < 8; ++t) acc[t] = zero;

    for (int kb = 0; kb < 128; ++kb) {
        unsigned int mrow = maskLds[ln * 128 + kb];
        Frag a;
#pragma unroll
        for (int v = 0; v < 8; ++v) {
            int k0 = k0A(v, hf);
            int j0 = kb * 32 + k0;
            float2 fdp = *(const float2*)&fdLds[h][j0];
            float w0 = ((mrow >> k0) & 1u)
                         ? __expf(lrelu(fsA + fdp.x) - rmax) * rinv : 0.f;
            float w1 = ((mrow >> (k0 + 1)) & 1u)
                         ? __expf(lrelu(fsA + fdp.y) - rmax) * rinv : 0.f;
            a.u[v] = pack2bf(w0, w1);
        }
#pragma unroll
        for (int t = 0; t < 8; ++t) {
            Frag b;
            const unsigned short* bp = Ht + (size_t)(hbase + t * 16 + ln) * N + kb * 32;
#pragma unroll
            for (int v = 0; v < 8; ++v)
                b.u[v] = *(const unsigned int*)(bp + k0B(v, hf));
            acc[t] = __builtin_amdgcn_wmma_f32_16x16x32_bf16(false, a.v, false, b.v,
                                                             (short)0, acc[t], false, false);
        }
    }

    // Epilogue: ELU, write f32 + bf16 copies (bf16 feeds layer-2 GEMM)
#pragma unroll
    for (int t = 0; t < 8; ++t) {
#pragma unroll
        for (int r = 0; r < 8; ++r) {
            int i = i0 + hf * 8 + r;
            int c = hbase + t * 16 + ln;
            float v = eluf(acc[t][r]);
            out_f32[(size_t)i * 1024 + c] = v;
            out_bf16[(size_t)i * 1024 + c] = f2bf(v);
        }
    }
}

// ---------------- layer 2 fused attention (1 head, Fout=16) + ELU + log_softmax ----------------
// One block = 16 rows; split-K across 8 waves, LDS reduce, then fused epilogue.
__global__ void __launch_bounds__(256)
attn2_kernel(const float* __restrict__ adj,
             const float* __restrict__ fsrc, const float* __restrict__ fdst,
             const unsigned short* __restrict__ Ht,  // h2^T bf16 [16][4096]
             float* __restrict__ out)                // [4096][16] log-softmax
{
    __shared__ float fdLds[4096];           // shared fdst staged via async copy (16 KB)
    __shared__ unsigned int maskLds[16 * 128];
    __shared__ float sMax[16];
    __shared__ float sInv[16];
    __shared__ float part[8 * 256];
    const int N = 4096;
    int lane = threadIdx.x & 31, wave = threadIdx.x >> 5;
    int hf = lane >> 4, ln = lane & 15;
    int i0 = blockIdx.x * 16;

    // Async staging of fdst (all 256 threads cooperate; overlaps with mask build)
    for (int j4 = threadIdx.x * 4; j4 < 4096; j4 += 1024) {
        async_g2l_b128(lds_addr32(&fdLds[j4]),
                       (unsigned long long)(size_t)(fdst + j4));
    }

    for (int wi = wave; wi < 2048; wi += 8) {
        int r = wi >> 7, cw = wi & 127;
        float v = adj[(size_t)(i0 + r) * N + cw * 32 + lane];
        unsigned long long b = __ballot(v > 0.f);
        if (lane == 0) maskLds[wi] = (unsigned int)b;
    }
    wait_asynccnt0();
    __syncthreads();

    for (int rr = 0; rr < 2; ++rr) {
        int r = wave * 2 + rr;
        float fs = fsrc[i0 + r];
        float m = GAT_NEG_INF, s = 0.f;
        for (int cw = 0; cw < 128; ++cw) {
            unsigned int mw = maskLds[r * 128 + cw];
            if (mw == 0u) continue;
            if ((mw >> lane) & 1u) {
                float e = lrelu(fs + fdLds[cw * 32 + lane]);
                float M = fmaxf(m, e);
                s = s * __expf(m - M) + __expf(e - M);
                m = M;
            }
        }
#pragma unroll
        for (int off = 16; off >= 1; off >>= 1) {
            float mo = __shfl_xor(m, off, 32);
            float so = __shfl_xor(s, off, 32);
            float M = fmaxf(m, mo);
            s = s * __expf(m - M) + so * __expf(mo - M);
            m = M;
        }
        if (lane == 0) {
            sMax[r] = m;
            sInv[r] = s > 0.f ? 1.f / s : 0.f;
        }
    }
    __syncthreads();

    float fsA = fsrc[i0 + ln];
    float rmax = sMax[ln], rinv = sInv[ln];
    v8f acc = {0.f, 0.f, 0.f, 0.f, 0.f, 0.f, 0.f, 0.f};

    for (int kb = wave; kb < 128; kb += 8) {   // uniform trip count per wave
        unsigned int mrow = maskLds[ln * 128 + kb];
        Frag a, b;
#pragma unroll
        for (int v = 0; v < 8; ++v) {
            int k0 = k0A(v, hf);
            int j0 = kb * 32 + k0;
            float2 fdp = *(const float2*)&fdLds[j0];
            float w0 = ((mrow >> k0) & 1u)
                         ? __expf(lrelu(fsA + fdp.x) - rmax) * rinv : 0.f;
            float w1 = ((mrow >> (k0 + 1)) & 1u)
                         ? __expf(lrelu(fsA + fdp.y) - rmax) * rinv : 0.f;
            a.u[v] = pack2bf(w0, w1);
        }
#pragma unroll
        for (int v = 0; v < 8; ++v)
            b.u[v] = *(const unsigned int*)(Ht + (size_t)ln * N + kb * 32 + k0B(v, hf));
        acc = __builtin_amdgcn_wmma_f32_16x16x32_bf16(false, a.v, false, b.v,
                                                      (short)0, acc, false, false);
    }

#pragma unroll
    for (int r = 0; r < 8; ++r)
        part[wave * 256 + (hf * 8 + r) * 16 + ln] = acc[r];
    __syncthreads();

    int t = threadIdx.x;
    int m = t >> 4, n = t & 15;
    float v = 0.f;
#pragma unroll
    for (int w = 0; w < 8; ++w) v += part[w * 256 + t];
    v = eluf(v);
    // log_softmax over the 16 features of row m (16 consecutive lanes per row)
    float mx = v;
#pragma unroll
    for (int off = 8; off >= 1; off >>= 1) mx = fmaxf(mx, __shfl_xor(mx, off, 16));
    float l = v - mx;
    float se = __expf(l);
#pragma unroll
    for (int off = 8; off >= 1; off >>= 1) se += __shfl_xor(se, off, 16);
    out[(size_t)(i0 + m) * 16 + n] = l - __logf(se);
}

extern "C" void kernel_launch(void* const* d_in, const int* in_sizes, int n_in,
                              void* d_out, int out_size, void* d_ws, size_t ws_size,
                              hipStream_t stream) {
    (void)in_sizes; (void)n_in; (void)out_size; (void)ws_size;
    const float* x      = (const float*)d_in[0];   // [4096,512]
    const float* adj    = (const float*)d_in[1];   // [4096,4096]
    const float* W1     = (const float*)d_in[2];   // [512,8,128] -> [512,1024]
    const float* a_src1 = (const float*)d_in[3];   // [8,128]
    const float* a_dst1 = (const float*)d_in[4];
    const float* W2     = (const float*)d_in[5];   // [1024,1,16] -> [1024,16]
    const float* a_src2 = (const float*)d_in[6];   // [16]
    const float* a_dst2 = (const float*)d_in[7];
    float* out = (float*)d_out;                    // [4096,16]

    char* ws = (char*)d_ws;
    size_t off = 0;
    auto alloc = [&](size_t bytes) -> void* {
        void* p = ws + off;
        off += (bytes + 255) & ~(size_t)255;
        return p;
    };
    unsigned short* xb    = (unsigned short*)alloc(4096ull * 512 * 2);   // x bf16
    unsigned short* w1t   = (unsigned short*)alloc(1024ull * 512 * 2);   // W1^T bf16
    float*          h1    = (float*)alloc(4096ull * 1024 * 4);           // x @ W1
    unsigned short* h1t   = (unsigned short*)alloc(4096ull * 1024 * 2);  // h1^T bf16
    float*          fsrc1 = (float*)alloc(8ull * 4096 * 4);
    float*          fdst1 = (float*)alloc(8ull * 4096 * 4);
    float*          h1out = (float*)alloc(4096ull * 1024 * 4);           // elu(attn@h1)
    unsigned short* x2b   = (unsigned short*)alloc(4096ull * 1024 * 2);  // bf16 of h1out
    unsigned short* w2t   = (unsigned short*)alloc(16ull * 1024 * 2);    // W2^T bf16
    float*          h2    = (float*)alloc(4096ull * 16 * 4);
    unsigned short* h2t   = (unsigned short*)alloc(16ull * 4096 * 2);    // h2^T bf16
    float*          fsrc2 = (float*)alloc(4096ull * 4);
    float*          fdst2 = (float*)alloc(4096ull * 4);

    // Layer 1
    cvt_bf16_kernel<<<(4096 * 512) / 256, 256, 0, stream>>>(x, xb, 4096 * 512);
    transpose_bf16_kernel<<<(512 * 1024) / 256, 256, 0, stream>>>(W1, w1t, 512, 1024);
    gemm_bf16_kernel<<<(256 * 64) / 8, 256, 0, stream>>>(xb, w1t, h1, h1t, 4096, 1024, 512);
    head_proj_kernel<<<(8 * 4096) / 256, 256, 0, stream>>>(h1, a_src1, a_dst1,
                                                           fsrc1, fdst1, 4096, 8, 128);
    attn1_kernel<<<256, 256, 0, stream>>>(adj, fsrc1, fdst1, h1t, h1out, x2b);

    // Layer 2
    transpose_bf16_kernel<<<(1024 * 16) / 256, 256, 0, stream>>>(W2, w2t, 1024, 16);
    gemm_bf16_kernel<<<(256 * 1) / 8, 256, 0, stream>>>(x2b, w2t, h2, h2t, 4096, 16, 1024);
    head_proj_kernel<<<4096 / 256, 256, 0, stream>>>(h2, a_src2, a_dst2,
                                                     fsrc2, fdst2, 4096, 1, 16);
    attn2_kernel<<<256, 256, 0, stream>>>(adj, fsrc2, fdst2, h2t, out);
}
